// NCDEEncoder_9405978378430
// MI455X (gfx1250) — matrix operand order repaired and checked by reading, not compiled, not run on hardware
//
#include <hip/hip_runtime.h>
#include <hip/hip_bf16.h>
#include <math.h>

// ---------------------------------------------------------------------------
// NCDE encoder for MI455X (gfx1250), wave32 + WMMA.
// Sequential Kutta-3/8 scan: one workgroup (8 waves), each wave owns one
// 16x16 (m,n) accumulator tile of z/k1..k4 for the whole 2047-step run.
// v2: layer loop fully unrolled (no scratch spill of weight fragments),
//     double-buffered LDS activations (1 barrier/layer), branchless native
//     tanh, software-pipelined w_out fragment loads.
// ---------------------------------------------------------------------------

typedef __attribute__((ext_vector_type(16))) _Float16 v16h;
typedef __attribute__((ext_vector_type(8)))  _Float16 v8h;
typedef __attribute__((ext_vector_type(8)))  float    v8f;

#define B_  32
#define T_  2048
#define F_  16
#define C_  17
#define H_  64

#define WINHID_FRAGS 32
#define WOUT_FRAGS   136
#define PACK_TOTAL   ((WINHID_FRAGS + WOUT_FRAGS) * 512)

#define WMMA_F16(A, Bf, Cf) \
  __builtin_amdgcn_wmma_f32_16x16x32_f16(false, (A), false, (Bf), (short)0, (Cf), false, false)

// branchless tanh on native v_exp/v_rcp: tanh(x)=sign(x)*(1-e)/(1+e), e=2^(-2*log2e*|x|)
__device__ __forceinline__ float fast_tanh(float x) {
  float ax = __builtin_fabsf(x);
  float e  = __builtin_amdgcn_exp2f(ax * -2.8853900817779268f);
  float r  = (1.0f - e) * __builtin_amdgcn_rcpf(1.0f + e);
  return __builtin_copysignf(r, x);
}

// ---------------------------------------------------------------------------
// A-fragment loader (16x32 f16, MxK) from an LDS [rows][64] f16 tile.
//   lane<16 : halves 0..7 -> K 0..7,   halves 8..15 -> K 8..15
//   lane>=16: halves 0..7 -> K 16..23, halves 8..15 -> K 24..31
// ---------------------------------------------------------------------------
__device__ __forceinline__ v16h load_afrag(const _Float16* base, int row0,
                                           int lane, int khalf) {
  int r   = row0 + (lane & 15);
  int off = (lane & 16) ? 16 : 0;
  const v8h* p0 = (const v8h*)(base + r * 64 + khalf * 32 + off);
  const v8h* p1 = (const v8h*)(base + r * 64 + khalf * 32 + 8 + off);
  v8h lo = *p0, hi = *p1;
  v16h out;
#pragma unroll
  for (int i = 0; i < 8; ++i) { out[i] = lo[i]; out[i + 8] = hi[i]; }
  return out;
}

// ---------------------------------------------------------------------------
// Kernel P: pre-pack weights into B-fragment order (32x16 f16, KxN).
//   lane<16 -> K = half_index (0..15); lane>=16 -> K = 16 + half_index.
// ---------------------------------------------------------------------------
__global__ void prepack_kernel(const float* __restrict__ w_in,
                               const float* __restrict__ w_hid,
                               const float* __restrict__ w_out,
                               _Float16* __restrict__ packed) {
  int gid = blockIdx.x * blockDim.x + threadIdx.x;
  if (gid >= PACK_TOTAL) return;
  const int WINHID = WINHID_FRAGS * 512;
  if (gid < WINHID) {
    int frag = gid >> 9, idx = gid & 511;
    int lane = idx >> 4, hi = idx & 15;
    int khalf = frag & 1;
    int n     = (frag >> 1) & 3;
    int layer = frag >> 3;
    int K   = khalf * 32 + ((lane >= 16) ? (hi + 16) : hi);
    int col = n * 16 + (lane & 15);
    const float* W = (layer == 0) ? w_in : (w_hid + (size_t)(layer - 1) * 64 * 64);
    packed[gid] = (_Float16)W[K * 64 + col];
  } else {
    int g = gid - WINHID;
    int frag = g >> 9, idx = g & 511;
    int lane = idx >> 4, hi = idx & 15;
    int khalf = frag & 1;
    int nc = frag >> 1;
    int n = nc / 17, c = nc % 17;
    int K = khalf * 32 + ((lane >= 16) ? (hi + 16) : hi);
    int h = n * 16 + (lane & 15);
    packed[gid] = (_Float16)w_out[(size_t)K * (C_ * H_) + h * C_ + c];
  }
}

// ---------------------------------------------------------------------------
// Kernel Z0: z0 = relu(xa0 @ wi1 + bi1) @ wi2 + bi2 -> z_T[:,0,:].
// ---------------------------------------------------------------------------
__global__ void z0_kernel(const float* __restrict__ x,
                          const float* __restrict__ wi1,
                          const float* __restrict__ bi1,
                          const float* __restrict__ wi2,
                          const float* __restrict__ bi2,
                          float* __restrict__ zT) {
  __shared__ float h1[B_ * 64];
  int tid = threadIdx.x;
  for (int e = tid; e < B_ * 64; e += 256) {
    int b = e >> 6, j = e & 63;
    float acc = bi1[j];                      // time channel xa0[0]=0
    for (int c = 1; c < C_; ++c)
      acc += x[(size_t)b * T_ * F_ + (c - 1)] * wi1[c * 64 + j];
    h1[e] = acc > 0.0f ? acc : 0.0f;
  }
  __syncthreads();
  for (int e = tid; e < B_ * 64; e += 256) {
    int b = e >> 6, j = e & 63;
    float acc = bi2[j];
    for (int k = 0; k < 64; ++k)
      acc += h1[b * 64 + k] * wi2[k * 64 + j];
    zT[(size_t)b * T_ * H_ + j] = acc;
  }
}

// ---------------------------------------------------------------------------
// Kernel S: sequential scan. 1 block, 256 threads = 8 waves.
// ---------------------------------------------------------------------------
__launch_bounds__(256, 1)
__global__ void scan_kernel(const float* __restrict__ x,
                            const float* __restrict__ b_in,
                            const float* __restrict__ b_hid,
                            const float* __restrict__ b_out,
                            const _Float16* __restrict__ packed,
                            float* __restrict__ zT) {
  __shared__ __align__(32) _Float16 uA0[B_ * 64];  // ping
  __shared__ __align__(32) _Float16 uA1[B_ * 64];  // pong
  __shared__ float xds[4][B_][C_];
  __shared__ float bias_l[4][64];
  __shared__ float bias_out[C_ * H_];

  const int tid  = threadIdx.x;
  const int lane = tid & 31;
  const int w    = tid >> 5;
  const int mt   = w >> 2;          // row tile 0..1
  const int nt   = w & 3;           // col tile 0..3
  const int lh   = lane >> 4;
  const int nl   = lane & 15;
  const int colh = nt * 16 + nl;    // this lane's output column

  for (int i = tid; i < 64; i += 256)       bias_l[0][i] = b_in[i];
  for (int i = tid; i < 3 * 64; i += 256)   bias_l[1 + (i >> 6)][i & 63] = b_hid[i];
  for (int i = tid; i < C_ * H_; i += 256)  bias_out[i] = b_out[i];

  // persistent layer-weight B fragments -- constant-indexed only (stay in VGPRs)
  v16h Bw[4][2];
#pragma unroll
  for (int layer = 0; layer < 4; ++layer)
#pragma unroll
    for (int kh = 0; kh < 2; ++kh) {
      int frag = (layer * 4 + nt) * 2 + kh;
      Bw[layer][kh] = *(const v16h*)(packed + (size_t)frag * 512 + lane * 16);
    }
  const _Float16* pwout = packed + WINHID_FRAGS * 512;

  // z fragment: row b = mt*16 + i + 8*lh, col h = colh
  v8f zf;
#pragma unroll
  for (int i = 0; i < 8; ++i) {
    int b = mt * 16 + i + 8 * lh;
    zf[i] = zT[(size_t)b * T_ * H_ + colh];
  }
  v8f k1f = {}, k2f = {}, k3f = {};
  __syncthreads();

  // one dense relu layer: read SRC buffer, 2 WMMAs, write DST buffer
#define DENSE_LAYER(SRC, DST, L)                                              \
  {                                                                           \
    v16h a0 = load_afrag(SRC, mt * 16, lane, 0);                              \
    v16h a1 = load_afrag(SRC, mt * 16, lane, 1);                              \
    v8f acc = {};                                                             \
    acc = WMMA_F16(a0, Bw[L][0], acc);                                        \
    acc = WMMA_F16(a1, Bw[L][1], acc);                                        \
    float bl = bias_l[L][colh];                                               \
    _Pragma("unroll")                                                         \
    for (int i = 0; i < 8; ++i) {                                             \
      float v = acc[i] + bl;                                                  \
      v = v > 0.0f ? v : 0.0f;                                                \
      int r = mt * 16 + i + 8 * lh;                                           \
      DST[r * 64 + colh] = (_Float16)v;                                       \
    }                                                                         \
    __syncthreads();                                                          \
  }

  for (int t = 0; t < T_ - 1; ++t) {
    // cubic-spline derivative samples:
    //   xd(0)=di, xd(1/3)=dlt, xd(2/3)=(4*dlt-di)/3, xd(1)=dlt
    for (int e = tid; e < B_ * C_; e += 256) {
      int b = e / C_, c = e % C_;
      float dlt, di;
      if (c == 0) { dlt = 1.0f; di = 1.0f; }
      else {
        const float* xb = x + (size_t)b * T_ * F_;
        float x0 = xb[t * F_ + (c - 1)];
        float x1 = xb[(t + 1) * F_ + (c - 1)];
        dlt = x1 - x0;
        di  = (t == 0) ? dlt : (x0 - xb[(t - 1) * F_ + (c - 1)]);
      }
      xds[0][b][c] = di;
      xds[1][b][c] = dlt;
      xds[2][b][c] = (4.0f * dlt - di) * (1.0f / 3.0f);
      xds[3][b][c] = dlt;
    }
    __syncthreads();

    v8f k4f = {};
#pragma unroll 1
    for (int s = 0; s < 4; ++s) {
      // RK stage input u = z + combo(k1..k3), all in matching fragment layout
      v8f u;
#pragma unroll
      for (int i = 0; i < 8; ++i) {
        float ki;
        if      (s == 0) ki = 0.0f;
        else if (s == 1) ki = k1f[i] * (1.0f / 3.0f);
        else if (s == 2) ki = k2f[i] - k1f[i] * (1.0f / 3.0f);
        else             ki = k1f[i] - k2f[i] + k3f[i];
        u[i] = zf[i] + ki;
      }
#pragma unroll
      for (int i = 0; i < 8; ++i) {
        int r = mt * 16 + i + 8 * lh;
        uA0[r * 64 + colh] = (_Float16)u[i];
      }
      __syncthreads();

      // 4 dense layers, ping-pong buffers, fully unrolled (Bw stays in VGPRs)
      DENSE_LAYER(uA0, uA1, 0)
      DENSE_LAYER(uA1, uA0, 1)
      DENSE_LAYER(uA0, uA1, 2)
      DENSE_LAYER(uA1, uA0, 3)

      // output layer sliced by c; tanh + einsum fold onto the fragment.
      // software-pipeline the w_out fragment loads (prefetch c+1).
      v16h h0 = load_afrag(uA0, mt * 16, lane, 0);
      v16h h1 = load_afrag(uA0, mt * 16, lane, 1);
      v8f kacc = {};
      v16h b0 = *(const v16h*)(pwout + (size_t)((nt * 17) * 2)     * 512 + lane * 16);
      v16h b1 = *(const v16h*)(pwout + (size_t)((nt * 17) * 2 + 1) * 512 + lane * 16);
#pragma unroll 1
      for (int c = 0; c < C_; ++c) {
        v16h p0 = b0, p1 = b1;
        if (c < C_ - 1) {
          int fb = (nt * 17 + c + 1) * 2;
          b0 = *(const v16h*)(pwout + (size_t)fb * 512 + lane * 16);
          b1 = *(const v16h*)(pwout + (size_t)(fb + 1) * 512 + lane * 16);
        }
        v8f y = {};
        y = WMMA_F16(h0, p0, y);
        y = WMMA_F16(h1, p1, y);
        float bo = bias_out[colh * C_ + c];
#pragma unroll
        for (int i = 0; i < 8; ++i) {
          int b = mt * 16 + i + 8 * lh;
          kacc[i] += fast_tanh(y[i] + bo) * xds[s][b][c];
        }
      }
      if      (s == 0) k1f = kacc;
      else if (s == 1) k2f = kacc;
      else if (s == 2) k3f = kacc;
      else             k4f = kacc;
      __syncthreads();          // h-frags consumed; next stage may reuse uA0
    }

    // Kutta 3/8 update + stream z_T
#pragma unroll
    for (int i = 0; i < 8; ++i) {
      zf[i] += (k1f[i] + 3.0f * (k2f[i] + k3f[i]) + k4f[i]) * 0.125f;
      int b = mt * 16 + i + 8 * lh;
      zT[(size_t)b * T_ * H_ + (size_t)(t + 1) * H_ + colh] = zf[i];
    }
  }
#undef DENSE_LAYER
}

// ---------------------------------------------------------------------------
// Kernel F: out = gelu(z_T) @ w_proj + b_proj (parallel WMMA) + mask zeros.
// ---------------------------------------------------------------------------
__launch_bounds__(256)
__global__ void proj_kernel(const float* __restrict__ zT,
                            const float* __restrict__ w_proj,
                            const float* __restrict__ b_proj,
                            float* __restrict__ out,
                            float* __restrict__ mask) {
  __shared__ __align__(32) _Float16 zg[128 * 64];
  __shared__ __align__(32) _Float16 pf[2 * 512];
  const int tid  = threadIdx.x;
  const int lane = tid & 31;
  const int w    = tid >> 5;
  const size_t r0 = (size_t)blockIdx.x * 128;

  for (int i = tid; i < 1024; i += 256) {
    int kh = i >> 9, idx = i & 511;
    int lane_i = idx >> 4, hi = idx & 15;
    int K = kh * 32 + ((lane_i >= 16) ? hi + 16 : hi);
    pf[i] = (_Float16)w_proj[K * 16 + (lane_i & 15)];
  }
  for (int i = tid; i < 128 * 64; i += 256) {
    float z = zT[r0 * 64 + i];
    float g = 0.5f * z * (1.0f + erff(z * 0.70710678118654752f));
    zg[i] = (_Float16)g;
  }
  __syncthreads();

  v16h a0 = load_afrag(zg, w * 16, lane, 0);
  v16h a1 = load_afrag(zg, w * 16, lane, 1);
  v16h b0 = *(const v16h*)(pf + lane * 16);
  v16h b1 = *(const v16h*)(pf + 512 + lane * 16);
  v8f acc = {};
  acc = WMMA_F16(a0, b0, acc);
  acc = WMMA_F16(a1, b1, acc);
  float bp = b_proj[lane & 15];
#pragma unroll
  for (int i = 0; i < 8; ++i) {
    size_t row = r0 + w * 16 + i + 8 * (lane >> 4);
    out[row * F_ + (lane & 15)] = acc[i] + bp;
  }
  if (tid < 128) mask[r0 + tid] = 0.0f;   // not_mask = all false
}

// ---------------------------------------------------------------------------
extern "C" void kernel_launch(void* const* d_in, const int* in_sizes, int n_in,
                              void* d_out, int out_size, void* d_ws, size_t ws_size,
                              hipStream_t stream) {
  const float* x      = (const float*)d_in[0];
  const float* wi1    = (const float*)d_in[1];
  const float* bi1    = (const float*)d_in[2];
  const float* wi2    = (const float*)d_in[3];
  const float* bi2    = (const float*)d_in[4];
  const float* w_in   = (const float*)d_in[5];
  const float* b_in   = (const float*)d_in[6];
  const float* w_hid  = (const float*)d_in[7];
  const float* b_hid  = (const float*)d_in[8];
  const float* w_out  = (const float*)d_in[9];
  const float* b_out  = (const float*)d_in[10];
  const float* w_proj = (const float*)d_in[11];
  const float* b_proj = (const float*)d_in[12];

  float* zT    = (float*)d_out;                      // (B, T, H)
  float* outp  = zT + (size_t)B_ * T_ * H_;          // (B, T, F)
  float* maskp = outp + (size_t)B_ * T_ * F_;        // (B, T) zeros

  _Float16* packed = (_Float16*)d_ws;

  prepack_kernel<<<(PACK_TOTAL + 255) / 256, 256, 0, stream>>>(w_in, w_hid, w_out, packed);
  z0_kernel<<<1, 256, 0, stream>>>(x, wi1, bi1, wi2, bi2, zT);
  scan_kernel<<<1, 256, 0, stream>>>(x, b_in, b_hid, b_out, packed, zT);
  proj_kernel<<<(B_ * T_) / 128, 256, 0, stream>>>(zT, w_proj, b_proj, outp, maskp);
}
